// SelfAttention_3745211482195
// MI455X (gfx1250) — compile-verified, matrix operand order checked by reading, not run on hardware
//
#include <hip/hip_runtime.h>
#include <cstdint>
#include <cstddef>

// ---------------------------------------------------------------------------
// Self-attention forward, N=8192, D=1024, fp32 in/out.
//   1) proj_gemm_kernel x3 : Qp=Q@Wq^T+bq (bf16), Kp=K@Wk^T+bk (bf16),
//                            VpT = (V@Wv^T+bv)^T (bf16, stored [D][N])
//   2) flash_attn_kernel   : online-softmax flash attention. K/V tiles are
//      staged into LDS by the Tensor Data Mover (tensor_load_to_lds, double
//      buffered, s_wait_tensorcnt), all GEMMs via v_wmma_f32_16x16x32_bf16.
// Workspace: 3 * N*D * 2 bytes = 48 MB of bf16 intermediates (fits in L2).
// ---------------------------------------------------------------------------

typedef unsigned short u16;
typedef unsigned int   u32;
typedef __attribute__((ext_vector_type(16))) __bf16 bf16x16;
typedef __attribute__((ext_vector_type(8)))  __bf16 bf16x8;
typedef __attribute__((ext_vector_type(8)))  u16    u16x8;
typedef __attribute__((ext_vector_type(8)))  float  f32x8;
typedef __attribute__((ext_vector_type(4)))  float  f32x4;
typedef __attribute__((ext_vector_type(4)))  u32    u32x4;
typedef __attribute__((ext_vector_type(8)))  int    i32x8;
typedef __attribute__((ext_vector_type(4)))  int    i32x4;

union Frag {
  bf16x16 v;
  bf16x8  h[2];
  u16x8   s[2];
  __bf16  e[16];
};

constexpr int N = 8192;
constexpr int D = 1024;

// flash-attention LDS staging geometry (TDM pads rows via OOB zero-fill)
constexpr int KLD  = 1032;              // K tile row stride (elems): 2064B, bank-spread
constexpr int VLD  = 40;                // V tile row stride (elems): 80B, bank-spread
constexpr int KBUF = 32 * KLD;          // one K buffer (elems)
constexpr int VBUF = 1024 * VLD;        // one V buffer (elems)
constexpr size_t DYN_LDS = (size_t)(2 * KBUF + 2 * VBUF) * sizeof(u16); // ~289KB

__device__ __forceinline__ u16 f2bf(float x) {
  return __builtin_bit_cast(u16, (__bf16)x);
}

__device__ __forceinline__ unsigned lds_off(const void* p) {
  // generic pointers to LDS carry the LDS byte offset in the low 32 bits
  return (unsigned)(size_t)p;
}

// ---------------------------------------------------------------------------
// Issue one 2-D TDM tile load: tile_d1 rows x tile_d0 elems (2-byte elems),
// row stride stride0 elems in memory, contiguous (tile_d0-stride) rows in LDS.
// tensor_d0/d1 bound the valid extent from the tile origin; tile_d0 > tensor_d0
// zero-fills the LDS row padding.  D# packing per CDNA5 ISA 8.3/8.4.
// ---------------------------------------------------------------------------
__device__ __forceinline__ void tdm_load_2d(unsigned lds_addr, const void* gptr,
                                            u32 tensor_d0, u32 tensor_d1,
                                            u32 tile_d0, u32 tile_d1,
                                            unsigned long long stride0) {
  unsigned long long ga = (unsigned long long)(size_t)gptr;
  u32x4 g0;
  g0[0] = 1u;                                  // count=1 (valid), user mode
  g0[1] = lds_addr;                            // LDS byte address
  g0[2] = (u32)ga;                             // global_addr[31:0]
  g0[3] = (u32)(ga >> 32) | (2u << 30);        // global_addr[56:32] | type=2
  i32x8 g1;
  g1[0] = (int)(1u << 16);                                       // data_size=2B
  g1[1] = (int)((tensor_d0 & 0xFFFFu) << 16);                    // dim0[15:0]
  g1[2] = (int)((tensor_d0 >> 16) | ((tensor_d1 & 0xFFFFu) << 16));
  g1[3] = (int)((tensor_d1 >> 16) | (tile_d0 << 16));            // tile_dim0
  g1[4] = (int)(tile_d1 & 0xFFFFu);                              // tile_dim1
  g1[5] = (int)(u32)(stride0 & 0xFFFFFFFFu);                     // dim0_stride lo
  g1[6] = (int)(u32)(stride0 >> 32);                             // dim0_stride hi
  g1[7] = 0;
  i32x4 z4 = {0, 0, 0, 0};
#if defined(__clang_major__) && (__clang_major__ >= 23)
  i32x8 z8 = {0, 0, 0, 0, 0, 0, 0, 0};
  __builtin_amdgcn_tensor_load_to_lds(g0, g1, z4, z4, z8, 0);
#else
  __builtin_amdgcn_tensor_load_to_lds(g0, g1, z4, z4, 0);
#endif
}

// ---------------------------------------------------------------------------
// Kernel 1: out[n,o] = sum_i X[n,i]*W[o,i] + b[o], written as bf16.
// Tile: 64(M) x 32(N) per workgroup, K-step 32, 8 waves (4x2 of 16x16 tiles).
// A/B staged through LDS with f32->bf16 conversion. transposed!=0 stores
// out^T (i.e. out[o*N + n]) with a contiguous b128 store per lane.
// ---------------------------------------------------------------------------
__global__ __launch_bounds__(256) void proj_gemm_kernel(
    const float* __restrict__ X, const float* __restrict__ W,
    const float* __restrict__ bias, u16* __restrict__ out, int transposed)
{
  __shared__ u16 Alds[64][40];   // 80B rows: 16B aligned, bank-spread
  __shared__ u16 Blds[32][40];

  const int tid  = threadIdx.x;
  const int lane = tid & 31;
  const int w    = tid >> 5;        // wave 0..7
  const int wr   = (w >> 1) * 16;   // wave row block inside 64-row tile
  const int wc   = (w & 1) * 16;    // wave col block inside 32-col tile
  const int m    = lane & 15;       // row (A) / col (B) index inside 16x16
  const int hi   = lane >> 4;       // k-half select
  const int kh   = hi * 8;

  const int r0 = blockIdx.x * 64;   // output rows (tokens)
  const int c0 = blockIdx.y * 32;   // output cols (d_model out)

  const int ar = tid >> 2;          // 0..63  A row
  const int ak = (tid & 3) * 8;     // 0,8,16,24
  const int br = tid >> 3;          // 0..31  B row (W out-row)
  const int bk = (tid & 7) * 4;     // 0..28

  f32x8 acc = {};

  for (int k0 = 0; k0 < D; k0 += 32) {
    f32x4 a0 = *(const f32x4*)&X[(size_t)(r0 + ar) * D + k0 + ak];
    f32x4 a1 = *(const f32x4*)&X[(size_t)(r0 + ar) * D + k0 + ak + 4];
    f32x4 b0 = *(const f32x4*)&W[(size_t)(c0 + br) * D + k0 + bk];
#pragma unroll
    for (int i = 0; i < 4; ++i) {
      Alds[ar][ak + i]     = f2bf(a0[i]);
      Alds[ar][ak + 4 + i] = f2bf(a1[i]);
      Blds[br][bk + i]     = f2bf(b0[i]);
    }
    __syncthreads();

    Frag a, b;
    a.s[0] = *(const u16x8*)&Alds[wr + m][kh];
    a.s[1] = *(const u16x8*)&Alds[wr + m][kh + 16];
    b.s[0] = *(const u16x8*)&Blds[wc + m][kh];
    b.s[1] = *(const u16x8*)&Blds[wc + m][kh + 16];
    acc = __builtin_amdgcn_wmma_f32_16x16x32_bf16(false, a.v, false, b.v,
                                                  (short)0, acc, false, false);
    __syncthreads();
  }

  const float bcol = bias[c0 + wc + m];
  if (!transposed) {
#pragma unroll
    for (int v = 0; v < 8; ++v) {
      const int row = r0 + wr + v + 8 * hi;
      out[(size_t)row * D + c0 + wc + m] = f2bf(acc[v] + bcol);
    }
  } else {
    union { u16x8 v8; u16 e[8]; } t;
#pragma unroll
    for (int v = 0; v < 8; ++v) t.e[v] = f2bf(acc[v] + bcol);
    *(u16x8*)&out[(size_t)(c0 + wc + m) * N + r0 + wr + 8 * hi] = t.v8;
  }
}

// ---------------------------------------------------------------------------
// Kernel 2: flash attention. One 16-query block per workgroup, 8 waves.
// Wave w owns output D-slice [w*128, w*128+128) (8 x 16x16 f32 accumulators).
// K/V tiles (32 keys x full D) are DMA'd into double-buffered LDS by the TDM
// (wave 0 issues; s_wait_tensorcnt 2 overlaps tile i+1's DMA with tile i's
// compute).  Per 32-key tile:
//   - each wave: partial S(16x32) over its 128-wide D slice (8 bf16 WMMAs,
//     operands from LDS), cross-wave reduce via ds_add_f32
//   - per-row online softmax (exp2), O-accumulator rescale
//   - P(bf16) @ V-tile : 8 bf16 WMMAs per wave from the LDS V buffer
// ---------------------------------------------------------------------------
__global__ __launch_bounds__(256) void flash_attn_kernel(
    const u16* __restrict__ Qb, const u16* __restrict__ Kb,
    const u16* __restrict__ VbT, float* __restrict__ Out)
{
  extern __shared__ u16 smem_dyn[];
  u16* Klds = smem_dyn;                 // [2][32][KLD]
  u16* Vlds = smem_dyn + 2 * KBUF;      // [2][1024][VLD]

  __shared__ float Slds[16][36];        // 144B rows: aligned, bank-spread
  __shared__ float CorrB[16];
  __shared__ float SumB[16];

  const int tid  = threadIdx.x;
  const int lane = tid & 31;
  const int w    = tid >> 5;
  const int m    = lane & 15;
  const int hi   = lane >> 4;
  const int kh   = hi * 8;
  const int q0   = blockIdx.x * 16;     // query block
  const int dw   = w * 128;             // this wave's D slice

  const float scale = 0.03125f;         // 1/sqrt(1024)
  const float L2E   = 1.44269504088896f;

  // Q fragments for this wave's D slice: 4 K-steps of 32 (stay in VGPRs)
  Frag qf[4];
#pragma unroll
  for (int ks = 0; ks < 4; ++ks) {
    const u16* p = &Qb[(size_t)(q0 + m) * D + dw + ks * 32 + kh];
    qf[ks].s[0] = *(const u16x8*)p;
    qf[ks].s[1] = *(const u16x8*)(p + 16);
  }

  f32x8 Oacc[8];
  {
    f32x8 z = {};
#pragma unroll
    for (int t = 0; t < 8; ++t) Oacc[t] = z;
  }

  float run_m = -3.0e38f;
  float run_l = 0.0f;

  // prologue: TDM-load tile 0 into buffer 0 (wave 0 issues for the workgroup)
  if (w == 0) {
    tdm_load_2d(lds_off(Klds), Kb,  /*t_d0=*/D, /*t_d1=*/32,
                /*tile_d0=*/KLD, /*tile_d1=*/32, /*stride0=*/D);
    tdm_load_2d(lds_off(Vlds), VbT, /*t_d0=*/32, /*t_d1=*/1024,
                /*tile_d0=*/VLD, /*tile_d1=*/1024, /*stride0=*/N);
  }

  for (int n0 = 0; n0 < N; n0 += 32) {
    const int buf = (n0 >> 5) & 1;

    if (w == 0) {
      const int nn = n0 + 32;
      if (nn < N) {
        const int nb = buf ^ 1;
        tdm_load_2d(lds_off(Klds + nb * KBUF), Kb + (size_t)nn * D,
                    D, 32, KLD, 32, D);
        tdm_load_2d(lds_off(Vlds + nb * VBUF), VbT + nn,
                    32, 1024, VLD, 1024, N);
        __builtin_amdgcn_s_wait_tensorcnt(2);   // tile n0 resident
      } else {
        __builtin_amdgcn_s_wait_tensorcnt(0);
      }
    }

    // zero the shared S tile (incl. padding)
    for (int i = tid; i < 16 * 36; i += 256) ((float*)Slds)[i] = 0.0f;
    __syncthreads();   // TDM data + zeroed S visible to all waves

    // partial S = Q_slice @ K_slice^T, two 16x16 subtiles (keys 0-15, 16-31)
    const u16* kbase = Klds + buf * KBUF;
#pragma unroll
    for (int st = 0; st < 2; ++st) {
      f32x8 s = {};
#pragma unroll
      for (int ks = 0; ks < 4; ++ks) {
        Frag kf;
        const u16* p = kbase + (st * 16 + m) * KLD + dw + ks * 32 + kh;
        kf.s[0] = *(const u16x8*)p;
        kf.s[1] = *(const u16x8*)(p + 16);
        s = __builtin_amdgcn_wmma_f32_16x16x32_bf16(false, qf[ks].v, false, kf.v,
                                                    (short)0, s, false, false);
      }
#pragma unroll
      for (int v = 0; v < 8; ++v)
        atomicAdd(&Slds[v + 8 * hi][st * 16 + m], s[v]);
    }
    __syncthreads();   // S reduction complete

    // per-row online softmax (each lane handles row m; identical across waves)
    f32x4 r[8];
#pragma unroll
    for (int j = 0; j < 8; ++j)
      r[j] = *(const f32x4*)&Slds[m][4 * j] * scale;

    float smax = -3.0e38f;
#pragma unroll
    for (int j = 0; j < 8; ++j)
      smax = fmaxf(smax, fmaxf(fmaxf(r[j].x, r[j].y), fmaxf(r[j].z, r[j].w)));

    const float new_m = fmaxf(run_m, smax);
    const float corr  = exp2f((run_m - new_m) * L2E);
    float lsum = 0.0f;
#pragma unroll
    for (int j = 0; j < 8; ++j) {
      f32x4 pj;
      pj.x = exp2f((r[j].x - new_m) * L2E);
      pj.y = exp2f((r[j].y - new_m) * L2E);
      pj.z = exp2f((r[j].z - new_m) * L2E);
      pj.w = exp2f((r[j].w - new_m) * L2E);
      r[j] = pj;
      lsum += pj.x + pj.y + pj.z + pj.w;
    }
    run_l = run_l * corr + lsum;
    run_m = new_m;

    // broadcast per-row correction so each lane can rescale rows v, v+8
    if (tid < 16) CorrB[tid] = corr;
    __syncthreads();

    float cf[8];
#pragma unroll
    for (int v = 0; v < 8; ++v) cf[v] = CorrB[v + 8 * hi];
#pragma unroll
    for (int t = 0; t < 8; ++t)
#pragma unroll
      for (int v = 0; v < 8; ++v) Oacc[t][v] *= cf[v];

    // assemble P bf16 A-fragment (k = key index 0..31, A-layout k packing)
    f32x4 pa0, pa1, pb0, pb1;
    if (hi == 0) { pa0 = r[0]; pa1 = r[1]; pb0 = r[4]; pb1 = r[5]; }
    else         { pa0 = r[2]; pa1 = r[3]; pb0 = r[6]; pb1 = r[7]; }
    Frag pf;
    pf.e[0]  = (__bf16)pa0.x;  pf.e[1]  = (__bf16)pa0.y;
    pf.e[2]  = (__bf16)pa0.z;  pf.e[3]  = (__bf16)pa0.w;
    pf.e[4]  = (__bf16)pa1.x;  pf.e[5]  = (__bf16)pa1.y;
    pf.e[6]  = (__bf16)pa1.z;  pf.e[7]  = (__bf16)pa1.w;
    pf.e[8]  = (__bf16)pb0.x;  pf.e[9]  = (__bf16)pb0.y;
    pf.e[10] = (__bf16)pb0.z;  pf.e[11] = (__bf16)pb0.w;
    pf.e[12] = (__bf16)pb1.x;  pf.e[13] = (__bf16)pb1.y;
    pf.e[14] = (__bf16)pb1.z;  pf.e[15] = (__bf16)pb1.w;

    // O += P @ V-tile (B fragments from the TDM-staged transposed V buffer)
    const u16* vbase = Vlds + buf * VBUF;
#pragma unroll
    for (int t = 0; t < 8; ++t) {
      Frag vf;
      const u16* vp = vbase + (dw + t * 16 + m) * VLD + kh;
      vf.s[0] = *(const u16x8*)vp;
      vf.s[1] = *(const u16x8*)(vp + 16);
      Oacc[t] = __builtin_amdgcn_wmma_f32_16x16x32_bf16(false, pf.v, false, vf.v,
                                                        (short)0, Oacc[t], false, false);
    }
    __syncthreads();   // all reads of buf done before TDM overwrites buf^1 next iter
  }

  // finalize: divide by row sums, write fp32 output
  if (tid < 16) SumB[tid] = run_l;
  __syncthreads();
  float rl[8];
#pragma unroll
  for (int v = 0; v < 8; ++v) rl[v] = 1.0f / SumB[v + 8 * hi];
#pragma unroll
  for (int t = 0; t < 8; ++t)
#pragma unroll
    for (int v = 0; v < 8; ++v)
      Out[(size_t)(q0 + v + 8 * hi) * D + dw + t * 16 + m] = Oacc[t][v] * rl[v];
}

// ---------------------------------------------------------------------------
extern "C" void kernel_launch(void* const* d_in, const int* in_sizes, int n_in,
                              void* d_out, int out_size, void* d_ws, size_t ws_size,
                              hipStream_t stream) {
  const float* Q  = (const float*)d_in[0];
  const float* K  = (const float*)d_in[1];
  const float* V  = (const float*)d_in[2];
  const float* Wq = (const float*)d_in[3];
  const float* bq = (const float*)d_in[4];
  const float* Wk = (const float*)d_in[5];
  const float* bk = (const float*)d_in[6];
  const float* Wv = (const float*)d_in[7];
  const float* bv = (const float*)d_in[8];
  float* out = (float*)d_out;

  // bf16 intermediates in workspace (requires 48 MB)
  u16* Qb  = (u16*)d_ws;
  u16* Kb  = Qb + (size_t)N * D;
  u16* VbT = Kb + (size_t)N * D;

  dim3 pgrid(N / 64, D / 32, 1);
  proj_gemm_kernel<<<pgrid, dim3(256), 0, stream>>>(Q, Wq, bq, Qb, 0);
  proj_gemm_kernel<<<pgrid, dim3(256), 0, stream>>>(K, Wk, bk, Kb, 0);
  proj_gemm_kernel<<<pgrid, dim3(256), 0, stream>>>(V, Wv, bv, VbT, 1);
  flash_attn_kernel<<<dim3(N / 16), dim3(256), DYN_LDS, stream>>>(Qb, Kb, VbT, out);

  (void)in_sizes; (void)n_in; (void)out_size; (void)ws_size;
}
